// LSTMCTop_52853867544976
// MI455X (gfx1250) — compile-verified, matrix operand order checked by reading, not run on hardware
//
#include <hip/hip_runtime.h>
#include <hip/hip_bf16.h>

typedef __attribute__((ext_vector_type(16))) __bf16 v16bf;
typedef __attribute__((ext_vector_type(8)))  float  v8f;

#define B_   64
#define T_   1024
#define I_   512
#define H_   1024
#define G4_  4096        // 4*H
#define K_   1536        // I + H
#define NCHUNK 48        // K/32
#define XCHUNK 16        // I/32
#define HCHUNK 32        // H/32

static constexpr size_t WFRAG_ELEMS = (size_t)NCHUNK * 4 * 64 * 32 * 16;  //  6,291,456
static constexpr size_t XFRAG_ELEMS = (size_t)T_ * XCHUNK * 4 * 32 * 16;  // 33,554,432
static constexpr size_t HFRAG_ELEMS = (size_t)HCHUNK * 4 * 32 * 16;       //     65,536

static constexpr size_t OFF_WFRAG = 0;
static constexpr size_t OFF_XFRAG = OFF_WFRAG + WFRAG_ELEMS * 2;
static constexpr size_t OFF_HFRAG = OFF_XFRAG + XFRAG_ELEMS * 2;          // 2 ping-pong buffers
static constexpr size_t OFF_TG    = OFF_HFRAG + 2 * HFRAG_ELEMS * 2;
static constexpr size_t OFF_BSUM  = OFF_TG   + (size_t)B_ * H_ * 4;
static constexpr size_t OFF_CNT   = OFF_BSUM + (size_t)G4_ * 4;

// ---- 16-bit A/B fragment k-mapping (16x32 operand, wave32) ----------------
// lanes 0-15 hold K = {0..7} U {16..23}; lanes 16-31 hold K = {8..15} U {24..31}
__device__ __forceinline__ int kk_from_lane_e(int lane, int e) {
  return (lane < 16) ? ((e < 8) ? e : e + 8) : ((e < 8) ? e + 8 : e + 16);
}
__device__ __forceinline__ void lane_e_from_kk(int kk, int M, int& lane, int& e) {
  if      (kk <  8) { lane = M;      e = kk;      }
  else if (kk < 16) { lane = M + 16; e = kk - 8;  }
  else if (kk < 24) { lane = M;      e = kk - 8;  }
  else              { lane = M + 16; e = kk - 16; }
}

// ---- prologue: pack [Wi | Wh] -> bf16 B-fragment layout -------------------
// wfrag flat: (((chunk*4 + g)*64 + strip)*32 + lane)*16 + e
__global__ __launch_bounds__(256) void pack_w_kernel(const float* __restrict__ Wi,
                                                     const float* __restrict__ Wh,
                                                     __bf16* __restrict__ wfrag) {
  int idx   = blockIdx.x * 256 + threadIdx.x;   // 0 .. 6,291,455
  int e     = idx & 15;
  int lane  = (idx >> 4)  & 31;
  int strip = (idx >> 9)  & 63;
  int g     = (idx >> 15) & 3;
  int chunk =  idx >> 17;                        // 0..47
  int n = g * H_ + strip * 16 + (lane & 15);     // gate row of W
  int k = chunk * 32 + kk_from_lane_e(lane, e);  // 0..1535
  float w = (k < I_) ? Wi[(size_t)n * I_ + k] : Wh[(size_t)n * H_ + (k - I_)];
  wfrag[idx] = (__bf16)w;
}

// ---- prologue: pack x -> bf16 A-fragment layout (all timesteps) -----------
// xfrag flat: ((((t*16 + chunk)*4 + mtile)*32 + lane)*16 + e
__global__ __launch_bounds__(256) void pack_x_kernel(const float* __restrict__ x,
                                                     __bf16* __restrict__ xfrag) {
  int idx   = blockIdx.x * 256 + threadIdx.x;   // 0 .. 33,554,431
  int e     = idx & 15;
  int lane  = (idx >> 4)  & 31;
  int mtile = (idx >> 9)  & 3;
  int chunk = (idx >> 11) & 15;
  int t     =  idx >> 15;
  int b = mtile * 16 + (lane & 15);
  int k = chunk * 32 + kk_from_lane_e(lane, e);
  xfrag[idx] = (__bf16)x[((size_t)b * T_ + t) * I_ + k];
}

// ---- prologue: tg = topic@Wt.T + bt, bsum = bi+bh, c-state, h0 frags, cnt -
__global__ __launch_bounds__(256) void setup_kernel(const float* __restrict__ h0,
                                                    const float* __restrict__ c0,
                                                    const float* __restrict__ topic,
                                                    const float* __restrict__ Wt,
                                                    const float* __restrict__ bt,
                                                    const float* __restrict__ bi,
                                                    const float* __restrict__ bh,
                                                    float* __restrict__ tg,
                                                    float* __restrict__ bsum,
                                                    float* __restrict__ cstate,
                                                    __bf16* __restrict__ hfrag,
                                                    unsigned* __restrict__ cnt) {
  int idx = blockIdx.x * 256 + threadIdx.x;     // 0 .. 65,535
  if (idx == 0) *cnt = 0u;
  if (idx < G4_) bsum[idx] = bi[idx] + bh[idx];

  { // tg + c-state
    int b = idx >> 10, j = idx & 1023;
    float acc = bt[j];
    const float* tr = topic + (size_t)b * 256;
    const float* wr = Wt + (size_t)j * 256;
    for (int k = 0; k < 256; ++k) acc += tr[k] * wr[k];
    tg[idx] = acc;
    cstate[idx] = c0[idx];
  }
  { // h0 -> fragment buffer 0
    int e = idx & 15, lane = (idx >> 4) & 31, mtile = (idx >> 9) & 3, hch = idx >> 11;
    int b = mtile * 16 + (lane & 15);
    int j = hch * 32 + kk_from_lane_e(lane, e);
    hfrag[idx] = (__bf16)h0[(size_t)b * H_ + j];
  }
}

// ---- persistent fused LSTM kernel: 32 blocks x 256 threads (8 waves) ------
__global__ __launch_bounds__(256) void lstm_persistent_kernel(
    const __bf16* __restrict__ wfrag,
    const __bf16* __restrict__ xfrag,
    __bf16*       __restrict__ hfrag,     // 2 x HFRAG_ELEMS
    const float*  __restrict__ tg,        // [B,H]
    const float*  __restrict__ bsum,      // [4H]
    float*        __restrict__ out,       // [B,T,H]
    float*        __restrict__ hN,        // [B,H]
    float*        __restrict__ cstate,    // [B,H]  (== cN output region)
    unsigned*     __restrict__ bar) {
  __shared__ float lds_g[4][B_][32];      // 32 KB gate exchange

  const int tid  = threadIdx.x;
  const int lane = tid & 31;
  const int widx = tid >> 5;              // 0..7
  const int g    = widx & 3;              // gate owned by this wave
  const int slocal = widx >> 2;           // 0..1 strip-in-block
  const int strip  = blockIdx.x * 2 + slocal;   // 0..63 (16 h-cols each)

  // per-chunk B pointer (stride 4*64*32*16 = 131072 elems per chunk)
  const __bf16* wp0 = wfrag + ((size_t)g * 64 + strip) * 512 + (size_t)lane * 16;

#pragma unroll 1
  for (int t = 0; t < T_; ++t) {
    const __bf16* hbuf = hfrag + (size_t)(t & 1) * HFRAG_ELEMS;
    __bf16* hnext      = hfrag + (size_t)((t + 1) & 1) * HFRAG_ELEMS;

    v8f Cacc[4] = {};                     // 4 M-tiles (B=64), one gate
    const __bf16* wp = wp0;

    // --- phase 1: x contribution (chunks 0..15), branch-free pointer walk --
    const __bf16* xp = xfrag + (size_t)t * 32768 + (size_t)lane * 16;
#pragma unroll 4
    for (int chunk = 0; chunk < XCHUNK; ++chunk) {
      v16bf a0 = *(const v16bf*)(xp);
      v16bf a1 = *(const v16bf*)(xp + 512);
      v16bf a2 = *(const v16bf*)(xp + 1024);
      v16bf a3 = *(const v16bf*)(xp + 1536);
      v16bf bf = *(const v16bf*)(wp);
      __builtin_prefetch(wp + 131072, 0, 1);      // next chunk's weights (L2)
      Cacc[0] = __builtin_amdgcn_wmma_f32_16x16x32_bf16(false, a0, false, bf, (short)0, Cacc[0], false, false);
      Cacc[1] = __builtin_amdgcn_wmma_f32_16x16x32_bf16(false, a1, false, bf, (short)0, Cacc[1], false, false);
      Cacc[2] = __builtin_amdgcn_wmma_f32_16x16x32_bf16(false, a2, false, bf, (short)0, Cacc[2], false, false);
      Cacc[3] = __builtin_amdgcn_wmma_f32_16x16x32_bf16(false, a3, false, bf, (short)0, Cacc[3], false, false);
      xp += 2048;
      wp += 131072;
    }

    // --- phase 2: h contribution (chunks 16..47) ---------------------------
    const __bf16* hp = hbuf + (size_t)lane * 16;
#pragma unroll 4
    for (int chunk = 0; chunk < HCHUNK; ++chunk) {
      v16bf a0 = *(const v16bf*)(hp);
      v16bf a1 = *(const v16bf*)(hp + 512);
      v16bf a2 = *(const v16bf*)(hp + 1024);
      v16bf a3 = *(const v16bf*)(hp + 1536);
      v16bf bf = *(const v16bf*)(wp);
      __builtin_prefetch(wp + 131072, 0, 1);
      Cacc[0] = __builtin_amdgcn_wmma_f32_16x16x32_bf16(false, a0, false, bf, (short)0, Cacc[0], false, false);
      Cacc[1] = __builtin_amdgcn_wmma_f32_16x16x32_bf16(false, a1, false, bf, (short)0, Cacc[1], false, false);
      Cacc[2] = __builtin_amdgcn_wmma_f32_16x16x32_bf16(false, a2, false, bf, (short)0, Cacc[2], false, false);
      Cacc[3] = __builtin_amdgcn_wmma_f32_16x16x32_bf16(false, a3, false, bf, (short)0, Cacc[3], false, false);
      hp += 2048;
      wp += 131072;
    }

    // spill gate tiles to LDS: C/D layout -> lane n = lane&15, m = r + 8*(lane>>4)
    const int ncol = slocal * 16 + (lane & 15);
    const int mhi  = 8 * (lane >> 4);
#pragma unroll
    for (int mt = 0; mt < 4; ++mt)
#pragma unroll
      for (int r = 0; r < 8; ++r)
        lds_g[g][mt * 16 + r + mhi][ncol] = Cacc[mt][r];
    __syncthreads();

    // fused cell update: 64 rows x 32 cols per block, 8 cells/thread
#pragma unroll
    for (int e = 0; e < 8; ++e) {
      int cell = tid * 8 + e;             // 0..2047
      int b    = cell >> 5;
      int colb = cell & 31;
      int j    = blockIdx.x * 32 + colb;  // h column
      float tgv = tg[(b << 10) + j];
      float ig = lds_g[0][b][colb] + bsum[j]            + tgv;
      float fg = lds_g[1][b][colb] + bsum[H_ + j]       + tgv;
      float cg = lds_g[2][b][colb] + bsum[2 * H_ + j];
      float og = lds_g[3][b][colb] + bsum[3 * H_ + j];
      float in_g = 1.0f / (1.0f + __expf(-ig));
      float f_g  = 1.0f / (1.0f + __expf(-fg));
      float c_g  = tanhf(cg);
      float o_g  = 1.0f / (1.0f + __expf(-og));
      float cp = cstate[(b << 10) + j];
      float cy = f_g * cp + in_g * c_g;
      float hy = o_g * tanhf(cy);
      cstate[(b << 10) + j] = cy;
      out[((size_t)b * T_ + t) * H_ + j] = hy;
      if (t == T_ - 1) hN[(b << 10) + j] = hy;
      // scatter h into next step's bf16 A-fragment
      int mtile = b >> 4, M = b & 15;
      int hch = j >> 5, kk = j & 31, l2, e2;
      lane_e_from_kk(kk, M, l2, e2);
      hnext[(((size_t)hch * 4 + mtile) * 32 + l2) * 16 + e2] = (__bf16)hy;
    }

    // device-wide barrier (monotonic counter, reset by setup_kernel each launch)
    __threadfence();
    __syncthreads();
    if (tid == 0) {
      __hip_atomic_fetch_add(bar, 1u, __ATOMIC_RELEASE, __HIP_MEMORY_SCOPE_AGENT);
      unsigned target = (unsigned)(t + 1) * gridDim.x;
      while (__hip_atomic_load(bar, __ATOMIC_ACQUIRE, __HIP_MEMORY_SCOPE_AGENT) < target)
        __builtin_amdgcn_s_sleep(1);
    }
    __syncthreads();
    __threadfence();
  }
}

extern "C" void kernel_launch(void* const* d_in, const int* in_sizes, int n_in,
                              void* d_out, int out_size, void* d_ws, size_t ws_size,
                              hipStream_t stream) {
  const float* x     = (const float*)d_in[0];
  const float* h0    = (const float*)d_in[1];
  const float* c0    = (const float*)d_in[2];
  const float* topic = (const float*)d_in[3];
  const float* Wi    = (const float*)d_in[4];
  const float* bi    = (const float*)d_in[5];
  const float* Wh    = (const float*)d_in[6];
  const float* bh    = (const float*)d_in[7];
  const float* Wt    = (const float*)d_in[8];
  const float* bt    = (const float*)d_in[9];

  char* ws = (char*)d_ws;
  __bf16*   wfrag = (__bf16*)(ws + OFF_WFRAG);
  __bf16*   xfrag = (__bf16*)(ws + OFF_XFRAG);
  __bf16*   hfrag = (__bf16*)(ws + OFF_HFRAG);
  float*    tg    = (float*)(ws + OFF_TG);
  float*    bsum  = (float*)(ws + OFF_BSUM);
  unsigned* bar   = (unsigned*)(ws + OFF_CNT);

  float* out    = (float*)d_out;                                  // [B,T,H]
  float* hN     = out + (size_t)B_ * T_ * H_;                     // [B,H]
  float* cstate = hN + (size_t)B_ * H_;                           // [B,H] == cN

  pack_w_kernel<<<(int)(WFRAG_ELEMS / 256), 256, 0, stream>>>(Wi, Wh, wfrag);
  pack_x_kernel<<<(int)(XFRAG_ELEMS / 256), 256, 0, stream>>>(x, xfrag);
  setup_kernel<<<256, 256, 0, stream>>>(h0, c0, topic, Wt, bt, bi, bh,
                                        tg, bsum, cstate, hfrag, bar);
  lstm_persistent_kernel<<<32, 256, 0, stream>>>(wfrag, xfrag, hfrag, tg, bsum,
                                                 out, hN, cstate, bar);
}
